// AttentionBlock_27367531610984
// MI455X (gfx1250) — compile-verified, matrix operand order checked by reading, not run on hardware
//
#include <hip/hip_runtime.h>
#include <hip/hip_bf16.h>

typedef __attribute__((ext_vector_type(16))) __bf16 v16bf;
typedef __attribute__((ext_vector_type(8)))  float  v8f;

#define WMMA_BF16(a, b, c) \
  __builtin_amdgcn_wmma_f32_16x16x32_bf16(false, (a), false, (b), (short)0, (c), false, false)

// ---------------------------------------------------------------------------
// CDNA5 async copy: global -> LDS without VGPR round-trip (ASYNCcnt-tracked).
// Per ISA 10.x: each lane copies 16B from its global address to its LDS address.
// ---------------------------------------------------------------------------
__device__ __forceinline__ void async_b128(const void* gsrc, void* lds_dst) {
  unsigned lds = (unsigned)(unsigned long long)lds_dst;  // generic low 32b == LDS offset
  unsigned long long g = (unsigned long long)gsrc;
  asm volatile("global_load_async_to_lds_b128 %0, %1, off"
               :: "v"(lds), "v"(g) : "memory");
}
__device__ __forceinline__ void wait_async_le3() {
  asm volatile("s_wait_asynccnt 0x3" ::: "memory");
}
__device__ __forceinline__ void wait_async_0() {
  asm volatile("s_wait_asynccnt 0x0" ::: "memory");
}

// ---------------------------------------------------------------------------
// Fragment loader for 16-bit WMMA operands (ISA 7.12.2).
// A-matrix 16x32:  lane m = lane&15; VGPR v holds K = (v>>2)*16 + half*8 + (v&3)*2 (+0/1)
// B-matrix 32x16 mirrors A with n = lane&15 when memory is laid out [n][k].
// ---------------------------------------------------------------------------
__device__ __forceinline__ v16bf load_frag16(const __bf16* base, int ld,
                                             int outer_base, int k_base) {
  const int lane  = threadIdx.x & 31;
  const int outer = outer_base + (lane & 15);
  const int half  = (lane >> 4) & 1;
  v16bf r;
#pragma unroll
  for (int v = 0; v < 8; ++v) {
    const int k = k_base + (v >> 2) * 16 + half * 8 + (v & 3) * 2;
    const __bf16* p = base + (size_t)outer * ld + k;
    r[2 * v]     = p[0];
    r[2 * v + 1] = p[1];
  }
  return r;
}

// ---------------------------------------------------------------------------
__global__ void cvt_bf16_kernel(const float* __restrict__ src,
                                __bf16* __restrict__ dst, int n) {
  int i = blockIdx.x * blockDim.x + threadIdx.x;
  int stride = gridDim.x * blockDim.x;
  for (; i < n; i += stride) dst[i] = (__bf16)src[i];
}

__global__ void zero_f32_kernel(float4* __restrict__ p, size_t n4) {
  size_t i = (size_t)blockIdx.x * blockDim.x + threadIdx.x;
  size_t stride = (size_t)gridDim.x * blockDim.x;
  float4 z = {0.f, 0.f, 0.f, 0.f};
  for (; i < n4; i += stride) p[i] = z;
}

// ---------------------------------------------------------------------------
// C[M,1024] = A[M,1024](bf16) @ W[1024,1024]^T (bf16, row o holds k) + bias
// Block tile 128x64, BK=32, 8 waves (4 over M, 2 over N), each wave 32x32.
// Double-buffered LDS staging via async global->LDS copies (software pipeline).
// ---------------------------------------------------------------------------
__global__ __launch_bounds__(256) void gemm_bf16_kernel(
    const __bf16* __restrict__ A,   // [M,1024]
    const __bf16* __restrict__ Bm,  // [1024 out][1024 k]  (== W row-major)
    const float*  __restrict__ bias,
    float* __restrict__ C) {
  __shared__ __align__(16) __bf16 As[2][128 * 32];
  __shared__ __align__(16) __bf16 Bs[2][64 * 32];
  const int t     = threadIdx.x;
  const int wid   = t >> 5;
  const int waveM = wid & 3;
  const int waveN = wid >> 2;
  const int m0 = blockIdx.y * 128;
  const int n0 = blockIdx.x * 64;

  // per-thread staging coordinates (contiguous 16B chunks)
  const int aIdx = t * 16;                   // A: 32B per thread (2 x b128)
  const int aM = aIdx >> 5, aK = aIdx & 31;
  const int bIdx = t * 8;                    // B: 16B per thread (1 x b128)
  const int bN = bIdx >> 5, bK = bIdx & 31;

  v8f acc[2][2] = {};

  // stage 0
  {
    const __bf16* ga = A + (size_t)(m0 + aM) * 1024 + aK;
    async_b128(ga,     &As[0][aIdx]);
    async_b128(ga + 8, &As[0][aIdx + 8]);
    async_b128(Bm + (size_t)(n0 + bN) * 1024 + bK, &Bs[0][bIdx]);
  }

  for (int it = 0; it < 32; ++it) {
    const int cur = it & 1;
    if (it + 1 < 32) {                       // issue next stage, then wait cur
      const int k0n = (it + 1) * 32;
      const __bf16* ga = A + (size_t)(m0 + aM) * 1024 + k0n + aK;
      async_b128(ga,     &As[cur ^ 1][aIdx]);
      async_b128(ga + 8, &As[cur ^ 1][aIdx + 8]);
      async_b128(Bm + (size_t)(n0 + bN) * 1024 + k0n + bK, &Bs[cur ^ 1][bIdx]);
      wait_async_le3();                      // 3 outstanding = next stage only
    } else {
      wait_async_0();
    }
    __syncthreads();                         // cur tile visible to all waves

    v16bf aF[2], bF[2];
#pragma unroll
    for (int i = 0; i < 2; ++i) aF[i] = load_frag16(As[cur], 32, waveM * 32 + i * 16, 0);
#pragma unroll
    for (int j = 0; j < 2; ++j) bF[j] = load_frag16(Bs[cur], 32, waveN * 32 + j * 16, 0);
#pragma unroll
    for (int i = 0; i < 2; ++i)
#pragma unroll
      for (int j = 0; j < 2; ++j)
        acc[i][j] = WMMA_BF16(aF[i], bF[j], acc[i][j]);

    __syncthreads();                         // all waves done reading cur
  }

  const int lane = t & 31;
  const int nIdx = lane & 15;
  const int half = lane >> 4;
#pragma unroll
  for (int i = 0; i < 2; ++i)
#pragma unroll
    for (int j = 0; j < 2; ++j) {
      int col = n0 + waveN * 32 + j * 16 + nIdx;
      float b = bias[col];
#pragma unroll
      for (int v = 0; v < 8; ++v) {
        int row = m0 + waveM * 32 + i * 16 + v + 8 * half;
        C[(size_t)row * 1024 + col] = acc[i][j][v] + b;
      }
    }
}

// ---------------------------------------------------------------------------
// Per-(token, head) RMS norm over D=64 + relayout [N,1024] -> [B,H,S,64] bf16.
// ---------------------------------------------------------------------------
__global__ __launch_bounds__(256) void norm_relayout_kernel(
    const float* __restrict__ raw, const float* __restrict__ w,
    __bf16* __restrict__ out) {
  int gw   = blockIdx.x * 8 + (threadIdx.x >> 5);
  int lane = threadIdx.x & 31;
  int n = gw >> 4;
  int h = gw & 15;
  int b = n >> 11;
  int s = n & 2047;
  int d0 = lane, d1 = lane + 32;
  float x0 = raw[(size_t)n * 1024 + h * 64 + d0];
  float x1 = raw[(size_t)n * 1024 + h * 64 + d1];
  float ss = x0 * x0 + x1 * x1;
#pragma unroll
  for (int m = 16; m >= 1; m >>= 1) ss += __shfl_xor(ss, m, 32);
  float y0 = x0, y1 = x1;
  if (w) {
    float r = rsqrtf(ss * (1.0f / 64.0f) + 1e-6f);
    y0 = x0 * r * w[d0];
    y1 = x1 * r * w[d1];
  }
  size_t ob = ((size_t)(b * 16 + h) * 2048 + s) * 64;
  out[ob + d0] = (__bf16)y0;
  out[ob + d1] = (__bf16)y1;
}

// ---------------------------------------------------------------------------
// Sliding-window attention. One wave per 16-query tile. 2 waves / block.
// ---------------------------------------------------------------------------
__global__ __launch_bounds__(64) void attn_kernel(
    const __bf16* __restrict__ Q,   // [B,H,S,64]
    const __bf16* __restrict__ K,
    const __bf16* __restrict__ V,
    const unsigned char* __restrict__ am,  // [B,S]
    float* __restrict__ attn,       // [B,H,S,S]
    __bf16* __restrict__ ctx) {     // [B,S,1024]
  __shared__ __bf16 Ks[2][16 * 64];
  __shared__ __bf16 Ps[2][16 * 288];
  __shared__ __bf16 Vs[2][64 * 32];
  const int w    = threadIdx.x >> 5;
  const int lane = threadIdx.x & 31;
  const int gid  = blockIdx.x * 2 + w;
  const int qt = gid & 127;
  const int h  = (gid >> 7) & 15;
  const int b  = gid >> 11;
  const int i0 = qt * 16;
  const size_t head = ((size_t)(b * 16 + h)) * 2048 * 64;
  const int nIdx = lane & 15;
  const int half = lane >> 4;

  v16bf aQ0 = load_frag16(Q + head + (size_t)i0 * 64, 64, 0, 0);
  v16bf aQ1 = load_frag16(Q + head + (size_t)i0 * 64, 64, 0, 32);

#pragma unroll
  for (int e = 0; e < 8; ++e) {
    int idx = e * 32 + lane;
    Ps[w][(idx >> 4) * 288 + 272 + (idx & 15)] = (__bf16)0.0f;
  }

  v8f accS[17];
#pragma unroll
  for (int kt = 0; kt < 17; ++kt) {
    const int j0 = i0 - 256 + kt * 16;
    __syncthreads();
#pragma unroll
    for (int e = 0; e < 32; ++e) {
      int idx = e * 32 + lane;           // == jj*64 + d
      int jj = idx >> 6, d = idx & 63;
      int j = j0 + jj;
      __bf16 val = (__bf16)0.0f;
      if (j >= 0 && j < 2048) val = K[head + (size_t)j * 64 + d];
      Ks[w][idx] = val;
    }
    {                                    // prefetch next K tile (near caches)
      int jn = j0 + 16 + (lane >> 1);
      if (jn >= 0 && jn < 2048)
        __builtin_prefetch(K + head + (size_t)jn * 64 + (lane & 1) * 32, 0, 3);
    }
    __syncthreads();
    v16bf bK0 = load_frag16(Ks[w], 64, 0, 0);
    v16bf bK1 = load_frag16(Ks[w], 64, 0, 32);
    v8f s = {};
    s = WMMA_BF16(aQ0, bK0, s);
    s = WMMA_BF16(aQ1, bK1, s);
    const int j = j0 + nIdx;
    const bool jin = (j >= 0) && (j < 2048);
    const bool kOK = jin && (am[(size_t)b * 2048 + (jin ? j : 0)] != 0);
#pragma unroll
    for (int v = 0; v < 8; ++v) {
      int i = i0 + v + 8 * half;
      bool ok = kOK && (j <= i) && (i - j < 256) &&
                (am[(size_t)b * 2048 + i] != 0);
      accS[kt][v] = ok ? s[v] * 0.125f : -__builtin_inff();
    }
  }

  float rinv[8];
#pragma unroll
  for (int v = 0; v < 8; ++v) {
    float t = accS[0][v];
#pragma unroll
    for (int kt = 1; kt < 17; ++kt) t = fmaxf(t, accS[kt][v]);
#pragma unroll
    for (int m = 8; m >= 1; m >>= 1) t = fmaxf(t, __shfl_xor(t, m, 32));
    float sum = 0.0f;
#pragma unroll
    for (int kt = 0; kt < 17; ++kt) {
      float p = __expf(accS[kt][v] - t);
      accS[kt][v] = p;
      sum += p;
    }
#pragma unroll
    for (int m = 8; m >= 1; m >>= 1) sum += __shfl_xor(sum, m, 32);
    rinv[v] = 1.0f / sum;
  }

  const size_t attnBase = ((size_t)(b * 16 + h)) * 2048 * 2048;
#pragma unroll
  for (int kt = 0; kt < 17; ++kt) {
    int j = i0 - 256 + kt * 16 + nIdx;
#pragma unroll
    for (int v = 0; v < 8; ++v) {
      int i = i0 + v + 8 * half;
      float p = accS[kt][v] * rinv[v];
      Ps[w][(v + 8 * half) * 288 + kt * 16 + nIdx] = (__bf16)p;
      if (j >= 0 && j < 2048)
        attn[attnBase + (size_t)i * 2048 + j] = p;
    }
  }

  v8f accC[4] = {};
  for (int ch = 0; ch < 9; ++ch) {
    __syncthreads();
#pragma unroll
    for (int e = 0; e < 64; ++e) {
      int idx = e * 32 + lane;           // == d*32 + kk
      int d = idx >> 5, kk = idx & 31;
      int key = i0 - 256 + ch * 32 + kk;
      __bf16 val = (__bf16)0.0f;
      if (key >= 0 && key < 2048) val = V[head + (size_t)key * 64 + d];
      Vs[w][idx] = val;
    }
    __syncthreads();
    v16bf aP = load_frag16(Ps[w], 288, 0, ch * 32);
#pragma unroll
    for (int ns = 0; ns < 4; ++ns) {
      v16bf bV = load_frag16(Vs[w], 32, ns * 16, 0);
      accC[ns] = WMMA_BF16(aP, bV, accC[ns]);
    }
  }
#pragma unroll
  for (int ns = 0; ns < 4; ++ns)
#pragma unroll
    for (int v = 0; v < 8; ++v) {
      int i = i0 + v + 8 * half;
      int col = h * 64 + ns * 16 + nIdx;
      ctx[((size_t)(b * 2048 + i)) * 1024 + col] = (__bf16)accC[ns][v];
    }
}

// ---------------------------------------------------------------------------
extern "C" void kernel_launch(void* const* d_in, const int* in_sizes, int n_in,
                              void* d_out, int out_size, void* d_ws, size_t ws_size,
                              hipStream_t stream) {
  (void)in_sizes; (void)n_in; (void)out_size; (void)ws_size;
  const float* X  = (const float*)d_in[0];
  const unsigned char* am = (const unsigned char*)d_in[1];
  const float* Wq = (const float*)d_in[2];
  const float* bq = (const float*)d_in[3];
  const float* Wk = (const float*)d_in[4];
  const float* bk = (const float*)d_in[5];
  const float* Wv = (const float*)d_in[6];
  const float* bv = (const float*)d_in[7];
  const float* Wo = (const float*)d_in[8];
  const float* bo = (const float*)d_in[9];
  const float* qnw = (const float*)d_in[10];
  const float* knw = (const float*)d_in[11];

  char* ws = (char*)d_ws;
  __bf16* Xbf  = (__bf16*)ws;                            // 8 MB
  __bf16* Wqbf = (__bf16*)(ws + ((size_t)8 << 20));      // 4 x 2 MB
  __bf16* Wkbf = Wqbf + (1 << 20);
  __bf16* Wvbf = Wkbf + (1 << 20);
  __bf16* Wobf = Wvbf + (1 << 20);
  float* Qraw = (float*)(ws + ((size_t)16 << 20));       // 3 x 16 MB
  float* Kraw = Qraw + (size_t)4096 * 1024;
  float* Vraw = Kraw + (size_t)4096 * 1024;
  __bf16* Qbf = (__bf16*)(ws + ((size_t)64 << 20));      // 3 x 8 MB
  __bf16* Kbf = Qbf + (size_t)4096 * 1024;
  __bf16* Vbf = Kbf + (size_t)4096 * 1024;
  __bf16* Ctx = Vbf + (size_t)4096 * 1024;               // 8 MB, ends at 96 MB

  float* outp  = (float*)d_out;                          // [B,S,1024]
  float* attnp = outp + (size_t)4096 * 1024;             // [B,H,S,S]

  cvt_bf16_kernel<<<1024, 256, 0, stream>>>(X,  Xbf,  4096 * 1024);
  cvt_bf16_kernel<<<512,  256, 0, stream>>>(Wq, Wqbf, 1024 * 1024);
  cvt_bf16_kernel<<<512,  256, 0, stream>>>(Wk, Wkbf, 1024 * 1024);
  cvt_bf16_kernel<<<512,  256, 0, stream>>>(Wv, Wvbf, 1024 * 1024);
  cvt_bf16_kernel<<<512,  256, 0, stream>>>(Wo, Wobf, 1024 * 1024);

  dim3 gg(16, 32);  // N/64 x M/128
  gemm_bf16_kernel<<<gg, 256, 0, stream>>>(Xbf, Wqbf, bq, Qraw);
  gemm_bf16_kernel<<<gg, 256, 0, stream>>>(Xbf, Wkbf, bk, Kraw);
  gemm_bf16_kernel<<<gg, 256, 0, stream>>>(Xbf, Wvbf, bv, Vraw);

  norm_relayout_kernel<<<8192, 256, 0, stream>>>(Qraw, qnw, Qbf);
  norm_relayout_kernel<<<8192, 256, 0, stream>>>(Kraw, knw, Kbf);
  norm_relayout_kernel<<<8192, 256, 0, stream>>>(Vraw, nullptr, Vbf);

  zero_f32_kernel<<<4096, 256, 0, stream>>>((float4*)attnp,
                                            ((size_t)2 * 16 * 2048 * 2048) / 4);
  attn_kernel<<<2048, 64, 0, stream>>>(Qbf, Kbf, Vbf, am, attnp, Ctx);

  gemm_bf16_kernel<<<gg, 256, 0, stream>>>(Ctx, Wobf, bo, outp);
}
